// DeformConv_38173669327479
// MI455X (gfx1250) — matrix operand order repaired and checked by reading, not compiled, not run on hardware
//
#include <hip/hip_runtime.h>

typedef __attribute__((ext_vector_type(16))) _Float16 v16h;
typedef __attribute__((ext_vector_type(4)))  _Float16 h4;
typedef __attribute__((ext_vector_type(8)))  float    v8f;
typedef __attribute__((ext_vector_type(4)))  float    f4;
typedef __attribute__((ext_vector_type(4)))  int      i4;

#define BATCH 4
#define CIN   256
#define HH    64
#define WW    64
#define HW    4096      // 64*64
#define K2    9
#define COUT  256
#define KTOT  2304      // CIN * K2
#define LSTR  264       // LDS row stride in halves: 528B, 16B-aligned, 4-bank lane stride

__device__ __forceinline__ int clamp63(int v) { return v < 0 ? 0 : (v > 63 ? 63 : v); }

// ---------------------------------------------------------------------------
// One-shot weight repack: f32 [Co, C, 3, 3] -> f16 [k][o][c] (tap-major,
// channel-contiguous). Turns the main kernel's 36B-strided weight reads into
// coalesced 8B f16 loads and hoists the fp32->f16 conversion out of the hot
// loop (done once instead of once per pixel-tile workgroup).
// ---------------------------------------------------------------------------
__global__ __launch_bounds__(256) void repack_wgt_kernel(
    const float* __restrict__ wgt,    // [Co, C, 3, 3]
    _Float16* __restrict__ wpk)       // [9, Co, C]
{
    const int c = threadIdx.x;        // 0..255
    const int o = blockIdx.x;         // 0..255
    const int k = blockIdx.y;         // 0..8
    wpk[((size_t)(k * COUT + o)) * CIN + c] =
        (_Float16)wgt[(size_t)o * KTOT + c * K2 + k];
}

// ---------------------------------------------------------------------------
// One workgroup: 64 (Co) x 64 (pixels) output tile for one batch image.
// 256 threads = 8 wave32s laid out 4(M) x 2(N); each wave owns 16x32 of the
// tile (two 16x16 f32 accumulators). Per 3x3 tap we stage the FULL K=256
// channel panel (weights + bilinear-sampled col) in LDS, then run 8 K-chunks
// = 16 WMMAs per wave between one barrier pair.
// ---------------------------------------------------------------------------
__global__ __launch_bounds__(256) void dcn_wmma_kernel(
    const float* __restrict__ x,      // [B, C, H, W]
    const float* __restrict__ off,    // [B, 18, Ho, Wo]
    const float* __restrict__ wgt,    // [Co, C, 3, 3]  (fallback path)
    const _Float16* __restrict__ wpk, // [9, Co, C]     (packed path)
    const int use_packed,
    float* __restrict__ out)          // [B, Co, Ho, Wo]
{
    // bilinear sampling metadata: 9 taps x 64 pixels, 4 corners each (~18.4KB)
    __shared__ alignas(16) int      sIdx[576 * 4];
    __shared__ alignas(16) float    sWgt[576 * 4];
    // full K-panel f16 tiles for one tap (~33.8KB each)
    __shared__ alignas(16) _Float16 wtile[64 * LSTR];   // [mo][ci]  (Co x 256)
    __shared__ alignas(16) _Float16 ctile[64 * LSTR];   // [p ][ci]  (N  x 256)

    const int tid = threadIdx.x;
    const int b   = blockIdx.z;
    const int m0  = blockIdx.y * 64;      // Co tile base
    const int n0  = blockIdx.x * 64;      // pixel tile base (one output row)

    // ---------------- phase 0: sampling metadata (once per tile) -----------
    for (int s = tid; s < 576; s += 256) {
        const int k = s >> 6;             // tap 0..8
        const int p = s & 63;             // pixel in tile
        const int n = n0 + p;
        const int oh = n >> 6, ow = n & 63;
        const float dy = off[((b * 18 + 2 * k    ) << 12) + n];
        const float dx = off[((b * 18 + 2 * k + 1) << 12) + n];
        const float py = dy + (float)(k / 3 + oh - 1);
        const float px = dx + (float)(k % 3 + ow - 1);
        const float y0f = floorf(py), x0f = floorf(px);
        const float ly = py - y0f, lx = px - x0f;
        const float hy = 1.0f - ly, hx = 1.0f - lx;
        const int y0 = (int)y0f, x0 = (int)x0f;
        const int y1 = y0 + 1,   x1 = x0 + 1;

        const int yy[4] = { y0, y0, y1, y1 };
        const int xx[4] = { x0, x1, x0, x1 };
        const float wc[4] = { hy * hx, hy * lx, ly * hx, ly * lx };
#pragma unroll
        for (int c = 0; c < 4; ++c) {
            const bool v = (yy[c] >= 0) & (yy[c] < HH) & (xx[c] >= 0) & (xx[c] < WW);
            sIdx[s * 4 + c] = (clamp63(yy[c]) << 6) | clamp63(xx[c]);
            sWgt[s * 4 + c] = v ? wc[c] : 0.0f;
        }
    }
    __syncthreads();

    // ---------------- wave / lane geometry ----------------------------------
    const int lane = tid & 31;
    const int wv   = tid >> 5;            // 0..7
    const int wm   = wv >> 1;             // 0..3  (M)
    const int wn   = wv & 1;              // 0..1  (N)
    const int lh   = lane & 15;
    const int hi16 = (lane >> 4) & 1;

    v8f acc0 = {};
    v8f acc1 = {};

    const float* __restrict__ xb = x + (size_t)b * CIN * HW;

    // ---------------- main loop over the 9 taps ------------------------------
    for (int k = 0; k < K2; ++k) {
        // ---- stage weight panel [64 Co x 256 ch] -> f16 LDS -----------------
        if (use_packed) {
            // coalesced 8B loads from the repacked f16 [k][o][c] layout
            const _Float16* __restrict__ wsrc =
                wpk + (size_t)(k * COUT + m0) * CIN;
#pragma unroll 4
            for (int i = 0; i < 16; ++i) {
                const int g  = i * 256 + tid;      // 0..4095
                const int q  = g & 63;             // channel group (ci = 4q)
                const int mo = g >> 6;             // Co row in tile
                *(h4*)&wtile[mo * LSTR + 4 * q] =
                    *(const h4*)&wsrc[(size_t)mo * CIN + 4 * q];
            }
        } else {
            // fallback: strided f32 loads + convert
#pragma unroll 4
            for (int i = 0; i < 16; ++i) {
                const int g  = i * 256 + tid;
                const int q  = g & 63;
                const int mo = g >> 6;
                const float* wp = &wgt[(size_t)(m0 + mo) * KTOT + (4 * q) * K2 + k];
                h4 hv;
                hv.x = (_Float16)wp[0 * K2];
                hv.y = (_Float16)wp[1 * K2];
                hv.z = (_Float16)wp[2 * K2];
                hv.w = (_Float16)wp[3 * K2];
                *(h4*)&wtile[mo * LSTR + 4 * q] = hv;
            }
        }

        // ---- stage bilinear-sampled col panel [64 px x 256 ch] -> f16 LDS ---
        // metadata (4 idx + 4 weights) amortized over 4 channels per group.
#pragma unroll 4
        for (int i = 0; i < 16; ++i) {
            const int g = i * 256 + tid;           // 0..4095
            const int q = g & 63;                  // channel group (ci = 4q)
            const int p = g >> 6;                  // pixel in tile
            const int s = (k << 6) + p;
            const i4 ix = *(const i4*)&sIdx[s * 4];
            const f4 wc = *(const f4*)&sWgt[s * 4];
            const float* __restrict__ xp = xb + (size_t)(4 * q) * HW;
            h4 hv;
#pragma unroll
            for (int c = 0; c < 4; ++c) {
                const float v = wc.x * xp[ix.x] + wc.y * xp[ix.y]
                              + wc.z * xp[ix.z] + wc.w * xp[ix.w];
                hv[c] = (_Float16)v;
                xp += HW;
            }
            *(h4*)&ctile[p * LSTR + 4 * q] = hv;
        }
        __syncthreads();

        // ---- 8 K-chunks of 32: 16 WMMAs per wave between barriers ----------
        const _Float16* wbase = &wtile[(wm * 16 + lh) * LSTR + (hi16 ? 16 : 0)];
        const _Float16* cbase = &ctile[(wn * 32 + lh) * LSTR + (hi16 ? 16 : 0)];
#pragma unroll
        for (int cc = 0; cc < CIN / 32; ++cc) {
            // A fragment (weights 16x32): lanes<16: K0-15 ; lanes>=16: K16-31
            v16h a;
            {
                f4* ap = (f4*)&a;
                const _Float16* wrow = wbase + cc * 32;
                ap[0] = *(const f4*)(wrow);
                ap[1] = *(const f4*)(wrow + 8);
            }
            // B fragments (col 32x16): lanes<16 hold K0-15, lanes>=16 hold K16-31
            const _Float16* crow0 = cbase + cc * 32;
            v16h bf0;
            {
                f4* bp = (f4*)&bf0;
                bp[0] = *(const f4*)(crow0);
                bp[1] = *(const f4*)(crow0 + 8);
            }
            acc0 = __builtin_amdgcn_wmma_f32_16x16x32_f16(
                false, a, false, bf0, (short)0, acc0, false, false);

            const _Float16* crow1 = crow0 + 16 * LSTR;
            v16h bf1;
            {
                f4* bp = (f4*)&bf1;
                bp[0] = *(const f4*)(crow1);
                bp[1] = *(const f4*)(crow1 + 8);
            }
            acc1 = __builtin_amdgcn_wmma_f32_16x16x32_f16(
                false, a, false, bf1, (short)0, acc1, false, false);
        }
        __syncthreads();
    }

    // ---------------- epilogue: scatter f32 accumulators ---------------------
    // C/D layout: VGPR i -> M = i (lanes<16) / i+8 (lanes>=16), N = lane%16
    const int mg  = m0 + wm * 16 + hi16 * 8;
    const int ng0 = n0 + wn * 32 + lh;
    float* __restrict__ ob = out + ((size_t)b * COUT << 12);
#pragma unroll
    for (int i = 0; i < 8; ++i) {
        ob[(size_t)(mg + i) * HW + ng0]      = acc0[i];
        ob[(size_t)(mg + i) * HW + ng0 + 16] = acc1[i];
    }
}

extern "C" void kernel_launch(void* const* d_in, const int* in_sizes, int n_in,
                              void* d_out, int out_size, void* d_ws, size_t ws_size,
                              hipStream_t stream) {
    (void)in_sizes; (void)n_in; (void)out_size;
    const float* x   = (const float*)d_in[0];   // [4,256,64,64]
    const float* off = (const float*)d_in[1];   // [4,18,64,64]
    const float* wgt = (const float*)d_in[2];   // [256,256,3,3]
    float* out = (float*)d_out;                 // [4,256,64,64]

    const size_t pack_bytes = (size_t)K2 * COUT * CIN * sizeof(_Float16); // ~1.18MB
    const int use_packed = (d_ws != nullptr && ws_size >= pack_bytes) ? 1 : 0;
    _Float16* wpk = (_Float16*)d_ws;

    if (use_packed) {
        repack_wgt_kernel<<<dim3(COUT, K2), 256, 0, stream>>>(wgt, wpk);
    }

    dim3 grid(64 /*pixel tiles*/, 4 /*Co tiles*/, 4 /*batch*/);
    dcn_wmma_kernel<<<grid, 256, 0, stream>>>(x, off, wgt, wpk, use_packed, out);
}
